// SDGraphEncoder_48284022341811
// MI455X (gfx1250) — matrix operand order repaired and checked by reading, not compiled, not run on hardware
//
#include <hip/hip_runtime.h>
#include <hip/hip_bf16.h>

// ---------------------------------------------------------------------------
// SDGraphEncoder for MI455X (gfx1250, wave32).
//   * All 1x1-conv MLP layers run as WMMA GEMMs: v_wmma_f32_16x16x32_f16
//     (f16 operands, f32 accumulate). 181-channel dims zero-padded to 192.
//   * Edge features (nbr-ctr || ctr) are built on the fly inside the GEMM
//     B-operand loader (no 160MB feat tensor).
//   * Dense path chunked per batch (16 chunks) to bound scratch.
//   * knn top-k, FPS, (1,3) convs: VALU kernels with LDS staging.
//
// Input order assumption (JAX pytree flatten, dict keys sorted):
//   0 sparse_fea[16,64,32] 1 dense_fea[16,64,2048] 2 stk_coor[16,32,32]
//   3..6   d2s  (w[64,64,1,3], b, g, bb)
//   7..14  dn1  (l0: w[181,256],b,g,bb ; l1: w[128,181],b,g,bb)
//   15..22 dn2   23..30 dn3
//   31..34 ds   (w[128,128,1,3], b, g, bb)
//   35..42 sp1  43..50 sp2  51..58 sp3
// Output: [union_sparse 16*128*16][union_dense 16*128*512][stk_coor_s 16*16*32]
// ---------------------------------------------------------------------------

typedef _Float16 h16;
typedef __attribute__((ext_vector_type(8)))  _Float16 v8h;
typedef __attribute__((ext_vector_type(16))) _Float16 v16h;
typedef __attribute__((ext_vector_type(8)))  float    v8f;

#define BN_RSQ 0.9999950000374996f   // 1/sqrt(1 + 1e-5)

__device__ __forceinline__ float gelu_f(float x) {
  float x3 = x * x * x;
  return 0.5f * x * (1.0f + tanhf(0.7978845608028654f * (x + 0.044715f * x3)));
}

// ---------------- weight convert: f32 [Or][Kr] -> f16 [Op][Kp], zero padded
__global__ void k_wconv(const float* __restrict__ src, h16* __restrict__ dst,
                        int Or, int Kr, int Op, int Kp) {
  int t = blockIdx.x * 256 + threadIdx.x;
  if (t >= Op * Kp) return;
  int o = t / Kp, kk = t - o * Kp;
  float v = (o < Or && kk < Kr) ? src[o * Kr + kk] : 0.0f;
  dst[t] = (h16)v;
}

// ---------------- DenseToSparse: conv2d(1,3) valid + BN + gelu + max over pts
// dense_fea [16,64,2048] viewed as [b][c][s=32][p=64] -> s_from_d [b][64][32]
__global__ void k_d2s(const float* __restrict__ dense, const float* __restrict__ w,
                      const float* __restrict__ bias, const float* __restrict__ g,
                      const float* __restrict__ bb, float* __restrict__ sfd) {
  __shared__ float tile[64][64];  // [c][p]
  int b = blockIdx.x >> 5, s = blockIdx.x & 31;
  for (int e = threadIdx.x; e < 4096; e += 64) {
    int c = e >> 6, p = e & 63;
    tile[c][p] = dense[((size_t)(b * 64 + c)) * 2048 + s * 64 + p];
  }
  __syncthreads();
  int o = threadIdx.x;
  float sc = g[o] * BN_RSQ, sh = bb[o], bo = bias[o];
  float m = -3.4e38f;
  for (int p = 0; p < 62; ++p) {
    float acc = bo;
    for (int c = 0; c < 64; ++c) {
      const float* wp = w + ((size_t)(o * 64 + c)) * 3;
      acc += tile[c][p] * wp[0] + tile[c][p + 1] * wp[1] + tile[c][p + 2] * wp[2];
    }
    m = fmaxf(m, gelu_f(acc * sc + sh));
  }
  sfd[((size_t)(b * 64 + o)) * 32 + s] = m;
}

// ---------------- build x0 rows (row-major [point][128]) --------------------
__global__ void k_build_x0_dn(const float* __restrict__ dense,
                              const float* __restrict__ sparse, float* __restrict__ x0) {
  size_t t = (size_t)blockIdx.x * 256 + threadIdx.x;
  if (t >= (size_t)32768 * 128) return;
  int c = (int)(t & 127);
  size_t bn = t >> 7;
  int n = (int)(bn & 2047), b = (int)(bn >> 11), s = n >> 6;
  x0[t] = (c < 64) ? dense[((size_t)(b * 64 + c)) * 2048 + n]
                   : sparse[((size_t)(b * 64 + (c - 64))) * 32 + s];
}
__global__ void k_build_x0_sp(const float* __restrict__ sparse,
                              const float* __restrict__ sfd, float* __restrict__ x0) {
  int t = blockIdx.x * 256 + threadIdx.x;
  if (t >= 512 * 128) return;
  int c = t & 127, bn = t >> 7, n = bn & 31, b = bn >> 5;
  x0[t] = (c < 64) ? sparse[(b * 64 + c) * 32 + n] : sfd[(b * 64 + (c - 64)) * 32 + n];
}

// ---------------- row sum-of-squares ---------------------------------------
__global__ void k_xx(const float* __restrict__ xr, float* __restrict__ xx, int nrows) {
  int r = blockIdx.x * 256 + threadIdx.x;
  if (r >= nrows) return;
  const float* p = xr + (size_t)r * 128;
  float s = 0.f;
  #pragma unroll 16
  for (int c = 0; c < 128; ++c) s += p[c] * p[c];
  xx[r] = s;
}

// ---------------- knn top-k (largest neg_d2, stable ties = lower index) ----
template <int KNN>
__global__ void k_knn(const float* __restrict__ xr, const float* __restrict__ xx,
                      int* __restrict__ idx, int Nper) {
  int bn = blockIdx.x * 64 + threadIdx.x;
  int b = bn / Nper, base = b * Nper;
  float own[128];
  #pragma unroll 16
  for (int c = 0; c < 128; ++c) own[c] = xr[(size_t)bn * 128 + c];
  float myxx = xx[bn];
  float tv[KNN]; int ti[KNN];
  #pragma unroll
  for (int j = 0; j < KNN; ++j) { tv[j] = -3.4e38f; ti[j] = base; }
  for (int m = 0; m < Nper; ++m) {
    const float* rm = xr + (size_t)(base + m) * 128;
    float dot = 0.f;
    #pragma unroll 16
    for (int c = 0; c < 128; ++c) dot += own[c] * rm[c];
    float v = 2.f * dot - myxx - xx[base + m];
    if (v > tv[KNN - 1]) {
      int j = KNN - 1;
      while (j > 0 && v > tv[j - 1]) { tv[j] = tv[j - 1]; ti[j] = ti[j - 1]; --j; }
      tv[j] = v; ti[j] = base + m;   // store GLOBAL row
    }
  }
  #pragma unroll
  for (int j = 0; j < KNN; ++j) idx[(size_t)bn * KNN + j] = ti[j];
}

// ---------------- WMMA GEMM: Y = act(bn(W x X + b)) ------------------------
// A: f16 [Opad][Kpad] row-major.  B (BMODE==0): f16 act [Plocal][Kpad] rows.
// B (BMODE==1): edge features built from f32 rows Xrows[.][128] + nbr idx.
// OUTMODE==0: write f16 [Plocal][Opad]; OUTMODE==1: write f32 [Plocal][Opad].
template <int BMODE, int OUTMODE>
__global__ void __launch_bounds__(256)
k_gemm(const h16* __restrict__ A, int Opad, int Oreal, int Kpad,
       const float* __restrict__ bias, const float* __restrict__ gamma,
       const float* __restrict__ beta, const h16* __restrict__ Bact,
       const float* __restrict__ Xrows, const int* __restrict__ nbr,
       int knn, long pbase, void* __restrict__ Out, int P) {
  int lane = threadIdx.x & 31;
  int wave = threadIdx.x >> 5;
  int tiles_o = Opad >> 4;
  long tile = (long)blockIdx.x * 8 + wave;
  long ntiles = (long)tiles_o * (P >> 4);
  if (tile >= ntiles) return;                  // uniform per wave
  int  o0 = (int)(tile % tiles_o) << 4;
  long p0 = (tile / tiles_o) << 4;
  int col = lane & 15;
  int hi  = lane >> 4;
  v8f acc = {};
  for (int k0 = 0; k0 < Kpad; k0 += 32) {
    // ---- A fragment: row = o0+col; 16 contiguous K starting at kb
    int kb = k0 + (hi << 4);
    const h16* ap = A + (size_t)(o0 + col) * Kpad + kb;
    if (k0 + 32 < Kpad) __builtin_prefetch(ap + 32, 0, 3);   // global_prefetch
    v8h alo = *(const v8h*)ap;
    v8h ahi = *(const v8h*)(ap + 8);
    v16h afrag;
    #pragma unroll
    for (int e = 0; e < 8; ++e) { afrag[e] = alo[e]; afrag[e + 8] = ahi[e]; }
    // ---- B fragment: col = lane&15, 16 contiguous K starting at kk
    int kk = k0 + (hi << 4);
    v16h bfrag;
    if (BMODE == 0) {
      bfrag = *(const v16h*)(Bact + (size_t)(p0 + col) * Kpad + kk);
    } else {
      long p  = pbase + p0 + col;
      long bn = p / knn;
      int  ng = nbr[p];
      if (kk < 128) {
        const float* rn = Xrows + (size_t)ng * 128 + kk;
        const float* rc = Xrows + (size_t)bn * 128 + kk;
        #pragma unroll
        for (int e = 0; e < 16; ++e) bfrag[e] = (h16)(rn[e] - rc[e]);
      } else {
        const float* rc = Xrows + (size_t)bn * 128 + (kk - 128);
        #pragma unroll
        for (int e = 0; e < 16; ++e) bfrag[e] = (h16)rc[e];
      }
    }
    acc = __builtin_amdgcn_wmma_f32_16x16x32_f16(false, afrag, false, bfrag,
                                                 (short)0, acc, false, false);
  }
  // ---- epilogue: elem j -> (M = j + 8*hi, N = col); bias + BN + gelu
  int mb = hi << 3;
  #pragma unroll
  for (int j = 0; j < 8; ++j) {
    int o = o0 + mb + j;
    float v = 0.0f;
    if (o < Oreal) {
      float z = acc[j] + bias[o];
      v = gelu_f(z * (gamma[o] * BN_RSQ) + beta[o]);
    }
    size_t oi = (size_t)(p0 + col) * Opad + o;
    if (OUTMODE == 0) ((h16*)Out)[oi] = (h16)v;
    else              ((float*)Out)[oi] = v;
  }
}

// ---------------- max over k neighbors -------------------------------------
__global__ void k_maxk(const float* __restrict__ y2, float* __restrict__ xout,
                       int knn, long bnbase) {
  int nloc = blockIdx.x, c = threadIdx.x;
  float m = -3.4e38f;
  for (int j = 0; j < knn; ++j)
    m = fmaxf(m, y2[((size_t)(nloc * knn + j)) * 128 + c]);
  xout[((size_t)(bnbase + nloc)) * 128 + c] = m;
}

// ---------------- concat x1||x2 -> f16 rows [P][256] -----------------------
__global__ void k_concat(const float* __restrict__ x1, const float* __restrict__ x2,
                         h16* __restrict__ x12) {
  size_t r = blockIdx.x;
  int t = threadIdx.x;
  float v = (t < 128) ? x1[r * 128 + t] : x2[r * 128 + (t - 128)];
  x12[r * 256 + t] = (h16)v;
}

// ---------------- FPS (N=32 -> 16) + write stk_coor_s ----------------------
__global__ void k_fps(const float* __restrict__ stk, int* __restrict__ fpsi,
                      float* __restrict__ out3) {
  __shared__ float dist[32];
  __shared__ float cen[32];
  __shared__ int   far_s;
  __shared__ int   fid[16];
  int b = blockIdx.x, tid = threadIdx.x;
  dist[tid] = 1e10f;
  if (tid == 0) far_s = 0;
  __syncthreads();
  for (int it = 0; it < 16; ++it) {
    int far = far_s;
    if (tid == 0) fid[it] = far;
    cen[tid] = stk[((size_t)(b * 32 + far)) * 32 + tid];
    __syncthreads();
    float d = 0.f;
    for (int c = 0; c < 32; ++c) {
      float u = stk[((size_t)(b * 32 + tid)) * 32 + c] - cen[c];
      d += u * u;
    }
    dist[tid] = fminf(dist[tid], d);
    __syncthreads();
    if (tid == 0) {
      float best = dist[0]; int bi = 0;
      for (int n = 1; n < 32; ++n) if (dist[n] > best) { best = dist[n]; bi = n; }
      far_s = bi;
    }
    __syncthreads();
  }
  if (tid < 16) fpsi[b * 16 + tid] = fid[tid];
  for (int sp = 0; sp < 16; ++sp)
    out3[((size_t)(b * 16 + sp)) * 32 + tid] = stk[((size_t)(b * 32 + fid[sp])) * 32 + tid];
}

// ---------------- gather sparse output -> d_out[0] [B][128][16] ------------
__global__ void k_gather_sp(const float* __restrict__ gsp, const int* __restrict__ fpsi,
                            float* __restrict__ out1) {
  int b = blockIdx.x >> 4, sp = blockIdx.x & 15, c = threadIdx.x;
  int fi = fpsi[b * 16 + sp];
  out1[((size_t)(b * 128 + c)) * 16 + sp] = gsp[((size_t)(b * 32 + fi)) * 128 + c];
}

// ---------------- downsample conv (1,3) stride(1,2) pad(0,1) + BN + gelu ---
__global__ void k_ds_conv(const float* __restrict__ gdn, const int* __restrict__ fpsi,
                          const float* __restrict__ w, const float* __restrict__ bias,
                          const float* __restrict__ g, const float* __restrict__ bb,
                          float* __restrict__ out2) {
  __shared__ float tile[64][128];  // [p][c]
  int b = blockIdx.x >> 4, sp = blockIdx.x & 15;
  int fi = fpsi[b * 16 + sp];
  size_t rowbase = (size_t)b * 2048 + (size_t)fi * 64;
  for (int e = threadIdx.x; e < 64 * 128; e += 128) {
    int p = e >> 7, c = e & 127;
    tile[p][c] = gdn[(rowbase + p) * 128 + c];
  }
  __syncthreads();
  int o = threadIdx.x;
  float sc = g[o] * BN_RSQ, sh = bb[o], bo = bias[o];
  for (int q = 0; q < 32; ++q) {
    float acc = bo;
    for (int t = 0; t < 3; ++t) {
      int pin = 2 * q + t - 1;
      if (pin < 0 || pin >= 64) continue;
      const float* wp = w + (size_t)o * 128 * 3 + t;
      for (int c = 0; c < 128; ++c) acc += tile[pin][c] * wp[c * 3];
    }
    out2[(((size_t)(b * 128 + o)) * 16 + sp) * 32 + q] = gelu_f(acc * sc + sh);
  }
}

// ===========================================================================
struct Mlp {
  const h16 *wa, *wb;
  const float *ba, *ga, *bba, *bb_, *gb, *bbb;
};

extern "C" void kernel_launch(void* const* d_in, const int* in_sizes, int n_in,
                              void* d_out, int out_size, void* d_ws, size_t ws_size,
                              hipStream_t stream) {
  (void)in_sizes; (void)n_in; (void)out_size;
  const float* sparse_fea = (const float*)d_in[0];
  const float* dense_fea  = (const float*)d_in[1];
  const float* stk_coor   = (const float*)d_in[2];

  // ---- scratch bump allocator ----
  unsigned char* wsp = (unsigned char*)d_ws;
  size_t off = 0;
  auto alloc = [&](size_t bytes) -> void* {
    void* p = wsp + off;
    off += (bytes + 255) & ~(size_t)255;
    return p;
  };
  float* x0dn = (float*)alloc((size_t)32768 * 128 * 4);
  float* x1dn = (float*)alloc((size_t)32768 * 128 * 4);
  float* x2dn = (float*)alloc((size_t)32768 * 128 * 4);
  float* gdn  = (float*)alloc((size_t)32768 * 128 * 4);
  float* xxb  = (float*)alloc((size_t)32768 * 4);
  int*   idxb = (int*)  alloc((size_t)32768 * 10 * 4);
  h16*   act1 = (h16*)  alloc((size_t)20480 * 192 * 2);
  float* y2   = (float*)alloc((size_t)20480 * 128 * 4);
  h16*   x12dn  = (h16*)alloc((size_t)32768 * 256 * 2);
  h16*   act3dn = (h16*)alloc((size_t)32768 * 192 * 2);
  float* x0sp = (float*)alloc((size_t)512 * 128 * 4);
  float* x1sp = (float*)alloc((size_t)512 * 128 * 4);
  float* x2sp = (float*)alloc((size_t)512 * 128 * 4);
  float* gsp  = (float*)alloc((size_t)512 * 128 * 4);
  float* xxsp = (float*)alloc((size_t)512 * 4);
  int*   idxsp = (int*) alloc((size_t)512 * 2 * 4);
  h16*   x12sp  = (h16*)alloc((size_t)512 * 256 * 2);
  h16*   act3sp = (h16*)alloc((size_t)512 * 192 * 2);
  float* sfd  = (float*)alloc((size_t)16 * 64 * 32 * 4);
  int*   fpsi = (int*)  alloc((size_t)256 * 4);
  h16* w16[12];
  for (int i = 0; i < 6; ++i) {
    w16[2 * i]     = (h16*)alloc((size_t)192 * 256 * 2);
    w16[2 * i + 1] = (h16*)alloc((size_t)128 * 192 * 2);
  }
  if (off > ws_size) return;  // workspace too small: deterministic no-op

  // ---- weight conversion + mlp param structs ----
  auto make_mlp = [&](int base, h16* wa, h16* wb) -> Mlp {
    k_wconv<<<(192 * 256 + 255) / 256, 256, 0, stream>>>((const float*)d_in[base + 0], wa, 181, 256, 192, 256);
    k_wconv<<<(128 * 192 + 255) / 256, 256, 0, stream>>>((const float*)d_in[base + 4], wb, 128, 181, 128, 192);
    Mlp m;
    m.wa = wa; m.ba = (const float*)d_in[base + 1]; m.ga = (const float*)d_in[base + 2]; m.bba = (const float*)d_in[base + 3];
    m.wb = wb; m.bb_ = (const float*)d_in[base + 5]; m.gb = (const float*)d_in[base + 6]; m.bbb = (const float*)d_in[base + 7];
    return m;
  };
  Mlp mdn1 = make_mlp(7,  w16[0], w16[1]);
  Mlp mdn2 = make_mlp(15, w16[2], w16[3]);
  Mlp mdn3 = make_mlp(23, w16[4], w16[5]);
  Mlp msp1 = make_mlp(35, w16[6], w16[7]);
  Mlp msp2 = make_mlp(43, w16[8], w16[9]);
  Mlp msp3 = make_mlp(51, w16[10], w16[11]);

  // ---- DenseToSparse + initial feature rows ----
  k_d2s<<<512, 64, 0, stream>>>(dense_fea, (const float*)d_in[3], (const float*)d_in[4],
                                (const float*)d_in[5], (const float*)d_in[6], sfd);
  k_build_x0_sp<<<(512 * 128 + 255) / 256, 256, 0, stream>>>(sparse_fea, sfd, x0sp);
  k_build_x0_dn<<<(int)(((size_t)32768 * 128 + 255) / 256), 256, 0, stream>>>(dense_fea, sparse_fea, x0dn);

  // ---- generic GCN runner ----
  auto run_gcn = [&](const float* x0, float* x1, float* x2, h16* x12, h16* act3,
                     float* gout, float* xxp, int* idxp, long totalRows, int Nper,
                     int knn, int ngroups, const Mlp& m1, const Mlp& m2, const Mlp& m3) {
    const int rowsPerGroup = (int)(totalRows / ngroups);
    const int Pg = rowsPerGroup * knn;
    auto launch_knn = [&](const float* xr) {
      k_xx<<<(int)((totalRows + 255) / 256), 256, 0, stream>>>(xr, xxp, (int)totalRows);
      int blocks = (int)(totalRows / 64);
      if (knn == 2) k_knn<2><<<blocks, 64, 0, stream>>>(xr, xxp, idxp, Nper);
      else          k_knn<10><<<blocks, 64, 0, stream>>>(xr, xxp, idxp, Nper);
    };
    auto edge_block = [&](const float* xin, const Mlp& m, float* xout) {
      launch_knn(xin);
      for (int g = 0; g < ngroups; ++g) {
        long pbase = (long)g * Pg;
        long nt1 = 12L * (Pg / 16);
        k_gemm<1, 0><<<(int)((nt1 + 7) / 8), 256, 0, stream>>>(
            m.wa, 192, 181, 256, m.ba, m.ga, m.bba, (const h16*)nullptr,
            xin, idxp, knn, pbase, act1, Pg);
        long nt2 = 8L * (Pg / 16);
        k_gemm<0, 1><<<(int)((nt2 + 7) / 8), 256, 0, stream>>>(
            m.wb, 128, 128, 192, m.bb_, m.gb, m.bbb, act1,
            (const float*)nullptr, (const int*)nullptr, 0, 0, y2, Pg);
        k_maxk<<<rowsPerGroup, 128, 0, stream>>>(y2, xout, knn, (long)g * rowsPerGroup);
      }
    };
    edge_block(x0, m1, x1);
    edge_block(x1, m2, x2);
    k_concat<<<(int)totalRows, 256, 0, stream>>>(x1, x2, x12);
    {
      long nt = 12L * (totalRows / 16);
      k_gemm<0, 0><<<(int)((nt + 7) / 8), 256, 0, stream>>>(
          m3.wa, 192, 181, 256, m3.ba, m3.ga, m3.bba, x12,
          (const float*)nullptr, (const int*)nullptr, 0, 0, act3, (int)totalRows);
    }
    {
      long nt = 8L * (totalRows / 16);
      k_gemm<0, 1><<<(int)((nt + 7) / 8), 256, 0, stream>>>(
          m3.wb, 128, 128, 192, m3.bb_, m3.gb, m3.bbb, act3,
          (const float*)nullptr, (const int*)nullptr, 0, 0, gout, (int)totalRows);
    }
  };

  // sparse: 512 rows, N=32, k=2, single chunk
  run_gcn(x0sp, x1sp, x2sp, x12sp, act3sp, gsp, xxsp, idxsp, 512, 32, 2, 1, msp1, msp2, msp3);
  // dense: 32768 rows, N=2048, k=10, 16 per-batch chunks
  run_gcn(x0dn, x1dn, x2dn, x12dn, act3dn, gdn, xxb, idxb, 32768, 2048, 10, 16, mdn1, mdn2, mdn3);

  // ---- downsample + outputs ----
  float* out  = (float*)d_out;
  float* out1 = out;                       // [16][128][16]
  float* out2 = out + 32768;               // [16][128][16][32]
  float* out3 = out + 32768 + 1048576;     // [16][16][32]
  k_fps<<<16, 32, 0, stream>>>(stk_coor, fpsi, out3);
  k_gather_sp<<<256, 128, 0, stream>>>(gsp, fpsi, out1);
  k_ds_conv<<<256, 128, 0, stream>>>(gdn, fpsi, (const float*)d_in[31],
                                     (const float*)d_in[32], (const float*)d_in[33],
                                     (const float*)d_in[34], out2);
}
